// MultiHeadAttention_30416958390778
// MI455X (gfx1250) — matrix operand order repaired and checked by reading, not compiled, not run on hardware
//
#include <hip/hip_runtime.h>

typedef _Float16 f16;
typedef __attribute__((ext_vector_type(16))) _Float16 v16h;
typedef __attribute__((ext_vector_type(8)))  _Float16 v8h;
typedef __attribute__((ext_vector_type(4)))  _Float16 v4h;
typedef __attribute__((ext_vector_type(8)))  float    v8f;
typedef __attribute__((ext_vector_type(4)))  float    v4f;

#define BATCH 2
#define HEADS 16
#define SEQ   2048
#define DM    1024
#define DEPTH 64
#define SSTRIDE 2056   // f32 score row stride (floats): 16B-aligned rows, rotates banks by 8/row
#define HSTRIDE 2056   // f16 prob row stride (halves): 8B-aligned rows

static __device__ __forceinline__ v16h cat8(v8h lo, v8h hi) {
    return __builtin_shufflevector(lo, hi, 0,1,2,3,4,5,6,7,8,9,10,11,12,13,14,15);
}

// ---------------------------------------------------------------------------
// Weight convert via LDS tile: Wt[n][k] = (f16) W[k][n]; coalesced both sides.
// ---------------------------------------------------------------------------
__global__ __launch_bounds__(256)
void mha_wt_transpose(const float* __restrict__ W, f16* __restrict__ Wt) {
    __shared__ f16 tile[32][33];
    const int bx = blockIdx.x & (DM / 32 - 1);  // k block
    const int by = blockIdx.x >> 5;             // n block
    const int tx = threadIdx.x & 31;
    const int ty = threadIdx.x >> 5;            // 8 rows per pass
    #pragma unroll
    for (int i = 0; i < 4; ++i)
        tile[ty + 8 * i][tx] = (f16)W[(size_t)(bx * 32 + ty + 8 * i) * DM + by * 32 + tx];
    __syncthreads();
    #pragma unroll
    for (int i = 0; i < 4; ++i)
        Wt[(size_t)(by * 32 + ty + 8 * i) * DM + bx * 32 + tx] = tile[tx][ty + 8 * i];
}

// ---------------------------------------------------------------------------
// Projection GEMM, register-blocked: each wave computes a 16x64 tile
// (1 A fragment reused by 4 B fragments -> 4 WMMA per A load; 128 WMMA/wave).
// VMODE 0 -> head-split [b][h][s][d]; VMODE 1 -> V transposed [b][h][d][s].
// ---------------------------------------------------------------------------
template <int VMODE>
__global__ __launch_bounds__(128)
void mha_qkv_proj(const float* __restrict__ X, const f16* __restrict__ Wt,
                  const float* __restrict__ bias, f16* __restrict__ out)
{
    const int wave = threadIdx.x >> 5;
    const int lane = threadIdx.x & 31;
    const int lm = lane & 15, lg = lane >> 4;

    const int job = blockIdx.x * 4 + wave;           // 256 m-tiles x 16 n-chunks
    const int mt = job >> 4;
    const int nc = job & 15;
    const int row0 = mt * 16, col0 = nc * 64;

    const float* aRow = X + (size_t)(row0 + lm) * DM;
    const f16* bRow0 = Wt + (size_t)(col0 +  0 + lm) * DM;
    const f16* bRow1 = Wt + (size_t)(col0 + 16 + lm) * DM;
    const f16* bRow2 = Wt + (size_t)(col0 + 32 + lm) * DM;
    const f16* bRow3 = Wt + (size_t)(col0 + 48 + lm) * DM;

    v8f c0 = {}, c1 = {}, c2 = {}, c3 = {};
    for (int kt = 0; kt < DM; kt += 32) {
        __builtin_prefetch(aRow + kt + 64, 0, 1);    // global_prefetch_b8
        v16h a;
        const int kb = kt + lg * 16;
        #pragma unroll
        for (int i = 0; i < 8; ++i) { a[i] = (f16)aRow[kb + i]; a[8 + i] = (f16)aRow[kb + 8 + i]; }
        const int ko = kt + lg * 16;                 // contiguous 32B B fragment
        v16h b0 = *(const v16h*)(bRow0 + ko);
        v16h b1 = *(const v16h*)(bRow1 + ko);
        v16h b2 = *(const v16h*)(bRow2 + ko);
        v16h b3 = *(const v16h*)(bRow3 + ko);
        c0 = __builtin_amdgcn_wmma_f32_16x16x32_f16(false, a, false, b0, (short)0, c0, false, false);
        c1 = __builtin_amdgcn_wmma_f32_16x16x32_f16(false, a, false, b1, (short)0, c1, false, false);
        c2 = __builtin_amdgcn_wmma_f32_16x16x32_f16(false, a, false, b2, (short)0, c2, false, false);
        c3 = __builtin_amdgcn_wmma_f32_16x16x32_f16(false, a, false, b3, (short)0, c3, false, false);
    }

    #pragma unroll
    for (int j = 0; j < 4; ++j) {
        const v8f c = (j == 0) ? c0 : (j == 1) ? c1 : (j == 2) ? c2 : c3;
        const int n = col0 + j * 16 + lm;
        const float bn = bias[n];
        const int h = n >> 6, d = n & (DEPTH - 1);
        #pragma unroll
        for (int i = 0; i < 8; ++i) {
            const int m  = row0 + i + 8 * lg;
            const int bb = m >> 11;
            const int s  = m & (SEQ - 1);
            const f16 val = (f16)(c[i] + bn);
            if (VMODE == 0)
                out[(((size_t)bb * HEADS + h) * SEQ + s) * DEPTH + d] = val;
            else
                out[(((size_t)bb * HEADS + h) * DEPTH + d) * SEQ + s] = val;
        }
    }
}

// ---------------------------------------------------------------------------
// Attention: one block (4 waves) per (b, h, 16 query rows).
//   P1: scores -> f32 LDS (WMMA, causal mask, scale); waves own contiguous ranges
//   P2: vectorized softmax (b128 LDS ops); writes f32 attn (coalesced 16B
//       global stores) + f16 probs to LDS (fragment-ready)
//   P3: ctx = probs @ v via WMMA (ds_load_b128 A fragments, zero converts)
// ---------------------------------------------------------------------------
__global__ __launch_bounds__(128)
void mha_attention(const f16* __restrict__ qh, const f16* __restrict__ kh,
                   const f16* __restrict__ vt, float* __restrict__ attn,
                   f16* __restrict__ ctx)
{
    __shared__ float sS[16 * SSTRIDE];     // 128.5 KB scores
    __shared__ f16   sP[16 * HSTRIDE];     // 64.3 KB normalized probs (f16)
    __shared__ float sRed[16][8];
    __shared__ float sStat[16];

    const int wave = threadIdx.x >> 5;
    const int lane = threadIdx.x & 31;
    const int lm = lane & 15, lg = lane >> 4;

    const int mt = blockIdx.x & 127;
    const int bh = blockIdx.x >> 7;
    const int b  = bh >> 4, h = bh & 15;
    const int row0 = mt * 16;

    const f16* Qb = qh + (size_t)bh * SEQ * DEPTH;
    const f16* Kb = kh + (size_t)bh * SEQ * DEPTH;
    const f16* Vb = vt + (size_t)bh * DEPTH * SEQ;

    // Q A-fragments for the two depth K-steps (constant for the whole block)
    const f16* qRow = Qb + (size_t)(row0 + lm) * DEPTH;
    const int qkb = lg * 16;
    v16h aq0 = cat8(*(const v8h*)(qRow + qkb),      *(const v8h*)(qRow + qkb + 8));
    v16h aq1 = cat8(*(const v8h*)(qRow + 32 + qkb), *(const v8h*)(qRow + 32 + qkb + 8));

    // ---- Phase 1: scores (each wave owns 32 contiguous key tiles) ----
    const float scale = 0.125f;
    for (int ntile = wave * 32; ntile < wave * 32 + 32; ++ntile) {
        const int colbase = ntile * 16;
        const f16* kRow = Kb + (size_t)(colbase + lm) * DEPTH;
        v16h b0 = *(const v16h*)(kRow + lg * 16);
        v16h b1 = *(const v16h*)(kRow + 32 + lg * 16);
        v8f c = {};
        c = __builtin_amdgcn_wmma_f32_16x16x32_f16(false, aq0, false, b0, (short)0, c, false, false);
        c = __builtin_amdgcn_wmma_f32_16x16x32_f16(false, aq1, false, b1, (short)0, c, false, false);
        #pragma unroll
        for (int i = 0; i < 8; ++i) {
            const int m = i + 8 * lg;
            const int col = colbase + lm;
            float v = c[i] * scale;
            if (col > row0 + m) v -= 1e9f;         // causal: mask * (-1e9)
            sS[m * SSTRIDE + col] = v;
        }
    }
    __syncthreads();

    // ---- Phase 2: vectorized softmax (thread j owns floats [32t+4j, +4)) ----
    {
        const int r = threadIdx.x >> 3;
        const int j = threadIdx.x & 7;
        float* rowp = sS + r * SSTRIDE + j * 4;
        f16*   prow = sP + r * HSTRIDE + j * 4;

        v4f vmx = { -3.4e38f, -3.4e38f, -3.4e38f, -3.4e38f };
        for (int t = 0; t < SEQ / 32; ++t)
            vmx = __builtin_elementwise_max(vmx, *(const v4f*)(rowp + t * 32));
        sRed[r][j] = fmaxf(fmaxf(vmx[0], vmx[1]), fmaxf(vmx[2], vmx[3]));
        __syncthreads();
        if (j == 0) {
            float m2 = sRed[r][0];
            #pragma unroll
            for (int q = 1; q < 8; ++q) m2 = fmaxf(m2, sRed[r][q]);
            sStat[r] = m2;
        }
        __syncthreads();
        const float rmax = sStat[r];
        v4f vsum = {};
        for (int t = 0; t < SEQ / 32; ++t) {
            v4f x = *(const v4f*)(rowp + t * 32);
            v4f e;
            #pragma unroll
            for (int q = 0; q < 4; ++q) e[q] = __expf(x[q] - rmax);
            *(v4f*)(rowp + t * 32) = e;
            vsum += e;
        }
        sRed[r][j] = vsum[0] + vsum[1] + vsum[2] + vsum[3];
        __syncthreads();
        if (j == 0) {
            float s2 = 0.f;
            #pragma unroll
            for (int q = 0; q < 8; ++q) s2 += sRed[r][q];
            sStat[r] = 1.0f / s2;
        }
        __syncthreads();
        const float rinv = sStat[r];
        float* gA = attn + ((size_t)bh * SEQ + row0 + r) * SEQ + j * 4;
        for (int t = 0; t < SEQ / 32; ++t) {
            v4f x = *(const v4f*)(rowp + t * 32);
            x *= rinv;
            *(v4f*)(gA + t * 32) = x;              // mandatory f32 attn (coalesced b128)
            v4h p;
            #pragma unroll
            for (int q = 0; q < 4; ++q) p[q] = (f16)x[q];
            *(v4h*)(prow + t * 32) = p;            // fragment-ready probs (b64)
        }
    }
    __syncthreads();

    // ---- Phase 3: ctx(16x64) = probs(16x2048) @ v(2048x64); wave owns 16 cols ----
    {
        v8f acc = {};
        const f16* vRow = Vb + (size_t)(wave * 16 + lm) * SEQ;
        const f16* sRow = sP + lm * HSTRIDE;
        for (int kt = 0; kt < SEQ; kt += 32) {
            const int kb = kt + lg * 16;
            v16h a = cat8(*(const v8h*)(sRow + kb), *(const v8h*)(sRow + kb + 8));
            v16h bf = *(const v16h*)(vRow + kt + lg * 16);
            acc = __builtin_amdgcn_wmma_f32_16x16x32_f16(false, a, false, bf, (short)0, acc, false, false);
        }
        #pragma unroll
        for (int i = 0; i < 8; ++i) {
            const int m = i + 8 * lg;
            const int srow = row0 + m;
            const int dcol = h * DEPTH + wave * 16 + lm;
            ctx[((size_t)b * SEQ + srow) * DM + dcol] = (f16)acc[i];
        }
    }
}

// ---------------------------------------------------------------------------
// Output projection, register-blocked 16x64/wave: out(f32) = ctx(f16)@Wo + bo
// ---------------------------------------------------------------------------
__global__ __launch_bounds__(128)
void mha_out_proj(const f16* __restrict__ Xh, const f16* __restrict__ Wt,
                  const float* __restrict__ bias, float* __restrict__ out)
{
    const int wave = threadIdx.x >> 5;
    const int lane = threadIdx.x & 31;
    const int lm = lane & 15, lg = lane >> 4;

    const int job = blockIdx.x * 4 + wave;
    const int mt = job >> 4;
    const int nc = job & 15;
    const int row0 = mt * 16, col0 = nc * 64;

    const f16* aRow  = Xh + (size_t)(row0 + lm) * DM;
    const f16* bRow0 = Wt + (size_t)(col0 +  0 + lm) * DM;
    const f16* bRow1 = Wt + (size_t)(col0 + 16 + lm) * DM;
    const f16* bRow2 = Wt + (size_t)(col0 + 32 + lm) * DM;
    const f16* bRow3 = Wt + (size_t)(col0 + 48 + lm) * DM;

    v8f c0 = {}, c1 = {}, c2 = {}, c3 = {};
    for (int kt = 0; kt < DM; kt += 32) {
        __builtin_prefetch(aRow + kt + 128, 0, 1);
        const int kb = kt + lg * 16;
        v16h a = cat8(*(const v8h*)(aRow + kb), *(const v8h*)(aRow + kb + 8));
        const int ko = kt + lg * 16;
        v16h b0 = *(const v16h*)(bRow0 + ko);
        v16h b1 = *(const v16h*)(bRow1 + ko);
        v16h b2 = *(const v16h*)(bRow2 + ko);
        v16h b3 = *(const v16h*)(bRow3 + ko);
        c0 = __builtin_amdgcn_wmma_f32_16x16x32_f16(false, a, false, b0, (short)0, c0, false, false);
        c1 = __builtin_amdgcn_wmma_f32_16x16x32_f16(false, a, false, b1, (short)0, c1, false, false);
        c2 = __builtin_amdgcn_wmma_f32_16x16x32_f16(false, a, false, b2, (short)0, c2, false, false);
        c3 = __builtin_amdgcn_wmma_f32_16x16x32_f16(false, a, false, b3, (short)0, c3, false, false);
    }

    #pragma unroll
    for (int j = 0; j < 4; ++j) {
        const v8f c = (j == 0) ? c0 : (j == 1) ? c1 : (j == 2) ? c2 : c3;
        const int n = col0 + j * 16 + lm;
        const float bn = bias[n];
        #pragma unroll
        for (int i = 0; i < 8; ++i) {
            const int m = row0 + i + 8 * lg;
            out[(size_t)m * DM + n] = c[i] + bn;
        }
    }
}

// ---------------------------------------------------------------------------
extern "C" void kernel_launch(void* const* d_in, const int* in_sizes, int n_in,
                              void* d_out, int out_size, void* d_ws, size_t ws_size,
                              hipStream_t stream)
{
    (void)in_sizes; (void)n_in; (void)out_size; (void)ws_size;

    const float* Q  = (const float*)d_in[0];
    const float* K  = (const float*)d_in[1];
    const float* V  = (const float*)d_in[2];
    /* d_in[3] = causal mask, applied analytically (triu k=1 -> -1e9) */
    const float* Wq = (const float*)d_in[4];
    const float* bq = (const float*)d_in[5];
    const float* Wk = (const float*)d_in[6];
    const float* bk = (const float*)d_in[7];
    const float* Wv = (const float*)d_in[8];
    const float* bv = (const float*)d_in[9];
    const float* Wo = (const float*)d_in[10];
    const float* bo = (const float*)d_in[11];

    float* out  = (float*)d_out;                             // [B,S,DM]
    float* attn = out + (size_t)BATCH * SEQ * DM;            // [B,H,S,S]

    char* ws = (char*)d_ws;
    const size_t wbytes = (size_t)DM * DM * sizeof(f16);
    const size_t hbytes = (size_t)BATCH * HEADS * SEQ * DEPTH * sizeof(f16);
    f16* WqT = (f16*)ws;              ws += wbytes;
    f16* WkT = (f16*)ws;              ws += wbytes;
    f16* WvT = (f16*)ws;              ws += wbytes;
    f16* WoT = (f16*)ws;              ws += wbytes;
    f16* qhd = (f16*)ws;              ws += hbytes;   // [b][h][s][d]
    f16* khd = (f16*)ws;              ws += hbytes;   // [b][h][s][d]
    f16* vhT = (f16*)ws;              ws += hbytes;   // [b][h][d][s]
    f16* ctx = (f16*)ws;              ws += hbytes;   // [b][s][dm]

    const int tblocks = (DM / 32) * (DM / 32);               // 1024
    mha_wt_transpose<<<tblocks, 256, 0, stream>>>(Wq, WqT);
    mha_wt_transpose<<<tblocks, 256, 0, stream>>>(Wk, WkT);
    mha_wt_transpose<<<tblocks, 256, 0, stream>>>(Wv, WvT);
    mha_wt_transpose<<<tblocks, 256, 0, stream>>>(Wo, WoT);

    const int gemmBlocks = (BATCH * SEQ / 16) * (DM / 64) / 4;   // 1024
    mha_qkv_proj<0><<<gemmBlocks, 128, 0, stream>>>(Q, WqT, bq, qhd);
    mha_qkv_proj<0><<<gemmBlocks, 128, 0, stream>>>(K, WkT, bk, khd);
    mha_qkv_proj<1><<<gemmBlocks, 128, 0, stream>>>(V, WvT, bv, vhT);

    mha_attention<<<BATCH * HEADS * (SEQ / 16), 128, 0, stream>>>(qhd, khd, vhT, attn, ctx);

    mha_out_proj<<<gemmBlocks, 128, 0, stream>>>(ctx, WoT, bo, out);
}
